// ExtractorCell_35150012350793
// MI455X (gfx1250) — matrix operand order repaired and checked by reading, not compiled, not run on hardware
//
#include <hip/hip_runtime.h>

typedef __attribute__((ext_vector_type(16))) __bf16 v16bf;
typedef __attribute__((ext_vector_type(8)))  float  v8f;

#define B_SZ 16384
#define I_SZ 512
#define H_SZ 512

// padded LDS row stride (in ushorts): 256 data + 8 pad -> 528B = 132 DWORDs (bank-conflict-free)
#define LROW 264

#if defined(__has_builtin)
#if __has_builtin(__builtin_amdgcn_tensor_load_to_lds)
#define HAVE_TDM 1
#endif
#endif

// ---------- helpers ----------
__device__ __forceinline__ unsigned short f2bf(float f) {
    union { float f; unsigned u; } x; x.f = f;
    unsigned u = x.u;
    unsigned r = u + 0x7FFFu + ((u >> 16) & 1u);   // round-to-nearest-even
    return (unsigned short)(r >> 16);
}
__device__ __forceinline__ float sigmoidf_(float x) { return 1.0f / (1.0f + __expf(-x)); }
__device__ __forceinline__ float tanhf_(float x) {
    float e = __expf(2.0f * x);                    // inf-safe: e=inf -> 1, e=0 -> -1
    return 1.0f - 2.0f / (e + 1.0f);
}

struct alignas(16) US8 { unsigned short v[8]; };

// ---------- prep: fp32 -> bf16 (8 elems/thread) ----------
__global__ void __launch_bounds__(256)
cvt_bf16_kernel(const float* __restrict__ src, unsigned short* __restrict__ dst, int n8) {
    int t = blockIdx.x * blockDim.x + threadIdx.x;
    if (t >= n8) return;
    const float4* s4 = (const float4*)src + (size_t)t * 2;
    float4 a = s4[0], b = s4[1];
    US8 o;
    o.v[0] = f2bf(a.x); o.v[1] = f2bf(a.y); o.v[2] = f2bf(a.z); o.v[3] = f2bf(a.w);
    o.v[4] = f2bf(b.x); o.v[5] = f2bf(b.y); o.v[6] = f2bf(b.z); o.v[7] = f2bf(b.w);
    *(uint4*)(dst + (size_t)t * 8) = *(uint4*)&o;
}

__global__ void __launch_bounds__(256)
scale_cvt_kernel(const float* __restrict__ s, const float* __restrict__ p,
                 unsigned short* __restrict__ dst, int n8) {
    int t = blockIdx.x * blockDim.x + threadIdx.x;
    if (t >= n8) return;
    int row = t / (I_SZ / 8);
    float pv = p[row];
    const float4* s4 = (const float4*)s + (size_t)t * 2;
    float4 a = s4[0], b = s4[1];
    US8 o;
    o.v[0] = f2bf(pv * a.x); o.v[1] = f2bf(pv * a.y); o.v[2] = f2bf(pv * a.z); o.v[3] = f2bf(pv * a.w);
    o.v[4] = f2bf(pv * b.x); o.v[5] = f2bf(pv * b.y); o.v[6] = f2bf(pv * b.z); o.v[7] = f2bf(pv * b.w);
    *(uint4*)(dst + (size_t)t * 8) = *(uint4*)&o;
}

#ifdef HAVE_TDM
typedef __attribute__((ext_vector_type(4))) unsigned int u32x4;
typedef __attribute__((ext_vector_type(8))) int          i32x8;
typedef __attribute__((ext_vector_type(4))) int          i32x4;

// TDM: load a 16-row x 256-elem bf16 tile (rows strided 512 elems in memory) into LDS,
// padding 16B after every 512B so LDS rows land on a 528B (conflict-free) stride.
__device__ __forceinline__ void tdm_load_rows(const unsigned short* gsrc, unsigned lds_byte_off) {
    unsigned long long ga = (unsigned long long)(uintptr_t)gsrc;
    u32x4 g0;
    g0[0] = 1u;                                          // count=1, user descriptor
    g0[1] = lds_byte_off;                                // lds_addr (bytes)
    g0[2] = (unsigned)(ga & 0xFFFFFFFFull);              // global_addr[31:0]
    g0[3] = (unsigned)((ga >> 32) & 0x01FFFFFFull) | (2u << 30);  // global_addr[56:32] | type=2

    i32x8 g1;
    g1[0] = (1 << 16)        // data_size = 2 bytes
          | (1 << 20)        // pad_enable
          | (6 << 22)        // pad_interval: 128 DWORDs (512B)
          | (3 << 25);       // pad_amount: 4 DWORDs (16B)
    g1[1] = (int)((256u & 0xFFFFu) << 16);               // tensor_dim0 = 256 (low 16)
    g1[2] = (int)(16u << 16);                            // tensor_dim0 hi=0 | tensor_dim1 = 16
    g1[3] = (int)(256u << 16);                           // tensor_dim1 hi=0 | tile_dim0 = 256
    g1[4] = 16;                                          // tile_dim1 = 16, tile_dim2 = 0
    g1[5] = 512;                                         // tensor_dim0_stride = 512 elems
    g1[6] = 0;
    g1[7] = 0;
    i32x4 z4 = {0, 0, 0, 0};                             // 2D tensor: groups 2/3 unused
    i32x8 z8 = {0, 0, 0, 0, 0, 0, 0, 0};                 // extra group (6-arg form): unused
    __builtin_amdgcn_tensor_load_to_lds(g0, g1, z4, z4, z8, 0);
}
#endif

// ---------- fused dual-GEMM + LSTM pointwise ----------
// gates[b,n] = sum_k act[b,k] * W[n,k]   (N = 4H, K = 512 twice)
// A bf16 16x32 layout: lane L holds row (L&15); K = kb + 8*(L>>4) + {0..7} and {16..23}
// B bf16 32x16 layout: lane L holds col (L&15); K = kb + 8*(L>>4) + {0..7} and {16..23}
// C/D f32:            lane L, elem v -> (M = v + 8*(L>>4), N = L&15)
__global__ void __launch_bounds__(256)
gates_lstm_kernel(const unsigned short* __restrict__ act1,   // [B,I] bf16 (p*s)
                  const unsigned short* __restrict__ act2,   // [B,H] bf16 (h_)
                  const unsigned short* __restrict__ wihb,   // [4H,I] bf16
                  const unsigned short* __restrict__ whhb,   // [4H,H] bf16
                  const float* __restrict__ b_ih, const float* __restrict__ b_hh,
                  const float* __restrict__ c_prev,          // [B,H]
                  float* __restrict__ h_out, float* __restrict__ c_out) {
    __shared__ __attribute__((aligned(16))) unsigned short smem[64 * LROW];  // ~33.8 KB

    const int lane = threadIdx.x & 31;
    const int wave = threadIdx.x >> 5;
    const int half = lane >> 4;
    const int l16  = lane & 15;

    const int rowBase = (blockIdx.x * 8 + wave) * 32;  // 32 batch rows per wave
    const int hBase   = blockIdx.y * 16;               // 16 hidden cols per block

    v8f acc[2][4];
#pragma unroll
    for (int t = 0; t < 2; ++t)
#pragma unroll
        for (int g = 0; g < 4; ++g)
#pragma unroll
            for (int v = 0; v < 8; ++v) acc[t][g][v] = 0.0f;

    const int aRow0 = rowBase + l16;
    const int aRow1 = rowBase + 16 + l16;

    // 4 K-chunks of 256: {W_ih k0-255, W_ih k256-511, W_hh k0-255, W_hh k256-511}
#pragma unroll
    for (int c = 0; c < 4; ++c) {
        const unsigned short* act = (c < 2) ? act1 : act2;
        const unsigned short* wgt = (c < 2) ? wihb : whhb;
        const int kbase = (c & 1) * 256;

        __syncthreads();  // previous chunk fully consumed before overwrite
#ifdef HAVE_TDM
        if (wave == 0) {
            const unsigned sbase = (unsigned)(uintptr_t)(&smem[0]);
#pragma unroll
            for (int g = 0; g < 4; ++g)
                tdm_load_rows(wgt + (size_t)(g * H_SZ + hBase) * 512 + kbase,
                              sbase + (unsigned)(g * 16 * LROW * 2));
            __builtin_amdgcn_s_wait_tensorcnt(0);
        }
#else
        // cooperative copy fallback, same padded layout: 64 rows x 32 x 16B
        for (int i = threadIdx.x; i < 2048; i += 256) {
            int r = i >> 5, q = i & 31;
            int g = r >> 4, rr = r & 15;
            uint4 v = *(const uint4*)(wgt + (size_t)(g * H_SZ + hBase + rr) * 512 + kbase + q * 8);
            *(uint4*)(&smem[r * LROW + q * 8]) = v;
        }
#endif
        __syncthreads();  // tile visible to all waves

#pragma unroll
        for (int kk = 0; kk < 256; kk += 32) {
            v16bf A0, A1;
            {
                const unsigned short* pa = act + (size_t)aRow0 * 512 + kbase + kk + 8 * half;
                ((uint4*)&A0)[0] = *(const uint4*)pa;
                ((uint4*)&A0)[1] = *(const uint4*)(pa + 16);
            }
            {
                const unsigned short* pa = act + (size_t)aRow1 * 512 + kbase + kk + 8 * half;
                ((uint4*)&A1)[0] = *(const uint4*)pa;
                ((uint4*)&A1)[1] = *(const uint4*)(pa + 16);
            }
#pragma unroll
            for (int g = 0; g < 4; ++g) {
                v16bf Bm;
                const unsigned short* pb = &smem[(g * 16 + l16) * LROW + kk + 8 * half];
                ((uint4*)&Bm)[0] = *(const uint4*)pb;     // ds_load_b128
                ((uint4*)&Bm)[1] = *(const uint4*)(pb + 16);
                acc[0][g] = __builtin_amdgcn_wmma_f32_16x16x32_bf16(
                    false, A0, false, Bm, (short)0, acc[0][g], false, false);
                acc[1][g] = __builtin_amdgcn_wmma_f32_16x16x32_bf16(
                    false, A1, false, Bm, (short)0, acc[1][g], false, false);
            }
        }
    }

    // per-lane column biases (column fixed per lane across all 8 acc elems)
    const int hcol = hBase + l16;
    const float bi  = b_ih[0 * H_SZ + hcol] + b_hh[0 * H_SZ + hcol];
    const float bff = b_ih[1 * H_SZ + hcol] + b_hh[1 * H_SZ + hcol];
    const float bg  = b_ih[2 * H_SZ + hcol] + b_hh[2 * H_SZ + hcol];
    const float bo  = b_ih[3 * H_SZ + hcol] + b_hh[3 * H_SZ + hcol];

#pragma unroll
    for (int t = 0; t < 2; ++t) {
#pragma unroll
        for (int v = 0; v < 8; ++v) {
            const int row = rowBase + t * 16 + 8 * half + v;
            const size_t idx = (size_t)row * H_SZ + hcol;
            float ig = sigmoidf_(acc[t][0][v] + bi);
            float fg = sigmoidf_(acc[t][1][v] + bff);
            float gg = tanhf_(acc[t][2][v] + bg);
            float og = sigmoidf_(acc[t][3][v] + bo);
            float cc = fg * c_prev[idx] + ig * gg;
            float hh = og * tanhf_(cc);
            c_out[idx] = cc;
            h_out[idx] = hh;
        }
    }
}

// ---------- p_new = sigmoid([h, h_new] . W_h2p + b) ; one wave per row ----------
__global__ void __launch_bounds__(256)
pnew_kernel(const float* __restrict__ h_prev, const float* __restrict__ h_new,
            const float* __restrict__ w, const float* __restrict__ b,
            float* __restrict__ p_out) {
    const int lane = threadIdx.x & 31;
    const int wave = threadIdx.x >> 5;
    const int row  = blockIdx.x * 8 + wave;

    const float4* hp = (const float4*)(h_prev + (size_t)row * H_SZ);
    const float4* hn = (const float4*)(h_new + (size_t)row * H_SZ);
    const float4* w0 = (const float4*)(w);
    const float4* w1 = (const float4*)(w + H_SZ);

    float sum = 0.0f;
#pragma unroll
    for (int it = 0; it < 4; ++it) {                 // 4 * 32 lanes * 4 floats = 512
        int j = it * 32 + lane;
        float4 a = hp[j], wa = w0[j];
        float4 c = hn[j], wc = w1[j];
        sum += a.x * wa.x + a.y * wa.y + a.z * wa.z + a.w * wa.w;
        sum += c.x * wc.x + c.y * wc.y + c.z * wc.z + c.w * wc.w;
    }
#pragma unroll
    for (int off = 16; off > 0; off >>= 1) sum += __shfl_xor(sum, off, 32);
    if (lane == 0) p_out[row] = sigmoidf_(sum + b[0]);
}

// ---------- launch ----------
extern "C" void kernel_launch(void* const* d_in, const int* in_sizes, int n_in,
                              void* d_out, int out_size, void* d_ws, size_t ws_size,
                              hipStream_t stream) {
    (void)in_sizes; (void)n_in; (void)out_size; (void)ws_size;
    const float* s     = (const float*)d_in[0];
    const float* h     = (const float*)d_in[1];
    const float* h_    = (const float*)d_in[2];
    const float* c_    = (const float*)d_in[3];
    const float* p     = (const float*)d_in[4];
    const float* W_ih  = (const float*)d_in[5];
    const float* W_hh  = (const float*)d_in[6];
    const float* b_ih  = (const float*)d_in[7];
    const float* b_hh  = (const float*)d_in[8];
    const float* W_h2p = (const float*)d_in[9];
    const float* b_h2p = (const float*)d_in[10];

    float* h_out = (float*)d_out;                       // [B,H]
    float* c_out = h_out + (size_t)B_SZ * H_SZ;         // [B,H]
    float* p_out = c_out + (size_t)B_SZ * H_SZ;         // [B]

    unsigned short* act1 = (unsigned short*)d_ws;                     // B*I
    unsigned short* act2 = act1 + (size_t)B_SZ * I_SZ;                // B*H
    unsigned short* wihb = act2 + (size_t)B_SZ * H_SZ;                // 4H*I
    unsigned short* whhb = wihb + (size_t)4 * H_SZ * I_SZ;            // 4H*H

    {   // bf16 conversions
        int n8 = B_SZ * I_SZ / 8;                                     // 1,048,576
        scale_cvt_kernel<<<n8 / 256, 256, 0, stream>>>(s, p, act1, n8);
        cvt_bf16_kernel<<<n8 / 256, 256, 0, stream>>>(h_, act2, n8);
        int w8 = 4 * H_SZ * I_SZ / 8;                                 // 131,072
        cvt_bf16_kernel<<<w8 / 256, 256, 0, stream>>>(W_ih, wihb, w8);
        cvt_bf16_kernel<<<w8 / 256, 256, 0, stream>>>(W_hh, whhb, w8);
    }

    {   // fused dual-GEMM + LSTM pointwise (TDM-staged weights in LDS)
        dim3 grid(B_SZ / (8 * 32), H_SZ / 16);                        // (64, 32)
        gates_lstm_kernel<<<grid, 256, 0, stream>>>(act1, act2, wihb, whhb,
                                                    b_ih, b_hh, c_, h_out, c_out);
    }

    {   // p_new
        pnew_kernel<<<B_SZ / 8, 256, 0, stream>>>(h, h_out, W_h2p, b_h2p, p_out);
    }
}
